// SignedGCN_1271310319653
// MI455X (gfx1250) — matrix operand order repaired and checked by reading, compiled, not run on hardware
//
#include <hip/hip_runtime.h>
#include <hip/hip_bf16.h>

#define N_NODES   12800
#define N_EDGES   409600
#define FEAT      200
#define HID       256
#define HALF      128
#define NUM_GRAPHS 64
#define NODE_SZ   200
#define KPAD      512              // padded K for layer GEMMs (2*256; 2*200 zero-padded)
#define RK        (NODE_SZ * HID)  // 51200 readout K

typedef __attribute__((ext_vector_type(16))) __bf16 v16bf;
typedef __attribute__((ext_vector_type(8)))  float  v8f;
typedef __attribute__((ext_vector_type(8)))  int    v8i_t;

__device__ __forceinline__ unsigned short f2bf(float f) {
    unsigned int u = __builtin_bit_cast(unsigned int, f);
    u += 0x7FFFu + ((u >> 16) & 1u);           // round-to-nearest-even
    return (unsigned short)(u >> 16);
}

__device__ __forceinline__ v16bf make_v16(uint4 a, uint4 b) {
    v8i_t t;
    t[0] = (int)a.x; t[1] = (int)a.y; t[2] = (int)a.z; t[3] = (int)a.w;
    t[4] = (int)b.x; t[5] = (int)b.y; t[6] = (int)b.z; t[7] = (int)b.w;
    return __builtin_bit_cast(v16bf, t);
}

// ---- f32 [O,K] -> bf16 [O,Kp] row-major, zero-padded ----
__global__ void pack_w_kernel(const float* __restrict__ W, unsigned short* __restrict__ out,
                              int O, int K, int Kp) {
    int tid = blockIdx.x * blockDim.x + threadIdx.x;
    if (tid >= O * Kp) return;
    int o = tid / Kp, k = tid - o * Kp;
    float v = (k < K) ? W[(long long)o * K + k] : 0.0f;
    out[tid] = f2bf(v);
}

// ---- signed degree counts ----
__global__ void deg_kernel(const int* __restrict__ ei, const float* __restrict__ w,
                           float* degp, float* degn) {
    int e = blockIdx.x * blockDim.x + threadIdx.x;
    if (e >= N_EDGES) return;
    float ww = w[e];
    int dst = ei[N_EDGES + e];
    if (ww > 0.f)      atomicAdd(&degp[dst], 1.f);
    else if (ww < 0.f) atomicAdd(&degn[dst], 1.f);
}

// ---- edge-weighted scatter: agg_{p,n}[dst] += |w| * cur[src], sign-split ----
__global__ void scatter_kernel(const float* __restrict__ cur, const int* __restrict__ ei,
                               const float* __restrict__ w,
                               float* aggp, float* aggn, int D, int chunks) {
    int gid = blockIdx.x * blockDim.x + threadIdx.x;
    if (gid >= N_EDGES * chunks) return;
    int e = gid / chunks;
    int c = (gid - e * chunks) * 4;
    float ww = w[e];
    if (ww == 0.f) return;
    int src = ei[e], dst = ei[N_EDGES + e];
    float4 v = *(const float4*)(cur + (long long)src * D + c);
    float s; float* agg;
    if (ww > 0.f) { s = ww;  agg = aggp; } else { s = -ww; agg = aggn; }
    float* p = agg + (long long)dst * HID + c;
    atomicAdd(p + 0, s * v.x); atomicAdd(p + 1, s * v.y);
    atomicAdd(p + 2, s * v.z); atomicAdd(p + 3, s * v.w);
}

// ---- build bf16 A matrices: [agg_p/deg_p | cur | 0] and [agg_n/deg_n | cur | 0], N x KPAD ----
__global__ void packA_kernel(const float* __restrict__ aggp, const float* __restrict__ aggn,
                             const float* __restrict__ degp, const float* __restrict__ degn,
                             const float* __restrict__ cur, int D,
                             unsigned short* __restrict__ Ap, unsigned short* __restrict__ An) {
    int tid = blockIdx.x * blockDim.x + threadIdx.x;     // N * 256 pairs
    if (tid >= N_NODES * (KPAD / 2)) return;
    int n  = tid >> 8;
    int k2 = tid & 255;
    int k  = k2 * 2;
    float p0, p1, q0, q1;
    if (k < D) {
        float ip  = 1.f / fmaxf(degp[n], 1.f);
        float in_ = 1.f / fmaxf(degn[n], 1.f);
        p0 = aggp[n * HID + k] * ip;  p1 = aggp[n * HID + k + 1] * ip;
        q0 = aggn[n * HID + k] * in_; q1 = aggn[n * HID + k + 1] * in_;
    } else if (k < 2 * D) {
        float c0 = cur[(long long)n * D + (k - D)];
        float c1 = cur[(long long)n * D + (k - D + 1)];
        p0 = q0 = c0; p1 = q1 = c1;
    } else {
        p0 = p1 = q0 = q1 = 0.f;
    }
    unsigned int up = (unsigned int)f2bf(p0) | ((unsigned int)f2bf(p1) << 16);
    unsigned int un = (unsigned int)f2bf(q0) | ((unsigned int)f2bf(q1) << 16);
    ((unsigned int*)Ap)[n * 256 + k2] = up;
    ((unsigned int*)An)[n * 256 + k2] = un;
}

// ---- per-layer dual-branch GEMM: h = act([Ap@Wp^T+bp || An@Wn^T+bn]) via bf16 WMMA ----
__global__ void gemm_layer_kernel(const unsigned short* __restrict__ Ap,
                                  const unsigned short* __restrict__ An,
                                  const unsigned short* __restrict__ Wp,
                                  const unsigned short* __restrict__ Wn,
                                  const float* __restrict__ bp, const float* __restrict__ bn,
                                  float* __restrict__ hout, unsigned short* __restrict__ hbf,
                                  int relu) {
    int lane  = threadIdx.x & 31;
    int wave  = threadIdx.x >> 5;
    int m0    = blockIdx.x * 16;
    int rowA  = m0 + (lane & 15);
    int kaoff = (lane >> 4) << 3;   // A: 0 or 8
    int kboff = (lane >> 4) << 3;   // B: 0 or 8
    for (int t = wave * 2; t < wave * 2 + 2; ++t) {      // 16 col-tiles over 8 waves
        int branch = t >> 3;
        int n0     = (t & 7) * 16;
        const unsigned short* A    = branch ? An : Ap;
        const unsigned short* W    = branch ? Wn : Wp;
        const float*          bias = branch ? bn : bp;
        const unsigned short* arow = A + rowA * KPAD;
        const unsigned short* brow = W + (n0 + (lane & 15)) * KPAD;
        v8f acc = {};
        for (int k0 = 0; k0 < KPAD; k0 += 32) {
            uint4 a0 = *(const uint4*)(arow + k0 + kaoff);
            uint4 a1 = *(const uint4*)(arow + k0 + kaoff + 16);
            uint4 b0 = *(const uint4*)(brow + k0 + kboff);
            uint4 b1 = *(const uint4*)(brow + k0 + kboff + 16);
            acc = __builtin_amdgcn_wmma_f32_16x16x32_bf16(
                false, make_v16(a0, a1), false, make_v16(b0, b1),
                (short)0, acc, false, false);
        }
        float bv    = bias[n0 + (lane & 15)];
        int   gcol  = branch * HALF + n0 + (lane & 15);
        int   mbase = m0 + ((lane >> 4) << 3);
        #pragma unroll
        for (int r = 0; r < 8; ++r) {
            float v = acc[r] + bv;
            if (relu) v = v > 0.f ? v : 0.01f * v;
            int m = mbase + r;
            hout[m * HID + gcol] = v;
            if (hbf) hbf[m * HID + gcol] = f2bf(v);
        }
    }
}

// ---- g init with readout bias ----
__global__ void ginit_kernel(const float* __restrict__ br, float* __restrict__ g) {
    int tid = blockIdx.x * blockDim.x + threadIdx.x;
    if (tid < NUM_GRAPHS * HID) g[tid] = br[tid & 255];
}

// ---- readout GEMM [64 x 51200] @ [51200 x 256], K-split with f32 atomic combine ----
__global__ void readout_gemm_kernel(const unsigned short* __restrict__ hbf,
                                    const unsigned short* __restrict__ Wr,
                                    float* __restrict__ g) {
    int lane = threadIdx.x & 31;
    int mi = blockIdx.x >> 4, ni = blockIdx.x & 15;
    int m0 = mi * 16, n0 = ni * 16;
    int kstart = blockIdx.y * 2048;
    const unsigned short* arow = hbf + (long long)(m0 + (lane & 15)) * RK;
    const unsigned short* brow = Wr  + (long long)(n0 + (lane & 15)) * RK;
    int kaoff = (lane >> 4) << 3, kboff = (lane >> 4) << 3;
    v8f acc = {};
    for (int k0 = kstart; k0 < kstart + 2048; k0 += 32) {
        uint4 a0 = *(const uint4*)(arow + k0 + kaoff);
        uint4 a1 = *(const uint4*)(arow + k0 + kaoff + 16);
        uint4 b0 = *(const uint4*)(brow + k0 + kboff);
        uint4 b1 = *(const uint4*)(brow + k0 + kboff + 16);
        acc = __builtin_amdgcn_wmma_f32_16x16x32_bf16(
            false, make_v16(a0, a1), false, make_v16(b0, b1),
            (short)0, acc, false, false);
    }
    int mbase = m0 + ((lane >> 4) << 3);
    int col   = n0 + (lane & 15);
    #pragma unroll
    for (int r = 0; r < 8; ++r)
        atomicAdd(&g[(mbase + r) * HID + col], acc[r]);
}

// ---- final [64 x 256] @ [256 x 1] + bl ----
__global__ void final_kernel(const float* __restrict__ g, const float* __restrict__ Wl,
                             const float* __restrict__ bl, float* __restrict__ out) {
    int i = threadIdx.x;
    if (i >= NUM_GRAPHS) return;
    float s = bl[0];
    for (int j = 0; j < HID; ++j) s += g[i * HID + j] * Wl[j];
    out[i] = s;
}

extern "C" void kernel_launch(void* const* d_in, const int* in_sizes, int n_in,
                              void* d_out, int out_size, void* d_ws, size_t ws_size,
                              hipStream_t stream) {
    (void)in_sizes; (void)n_in; (void)out_size; (void)ws_size;
    const float* x  = (const float*)d_in[0];
    const int*   ei = (const int*)d_in[1];
    const float* ew = (const float*)d_in[2];
    const float* Wp[3] = {(const float*)d_in[4],  (const float*)d_in[8],  (const float*)d_in[12]};
    const float* bp[3] = {(const float*)d_in[5],  (const float*)d_in[9],  (const float*)d_in[13]};
    const float* Wn[3] = {(const float*)d_in[6],  (const float*)d_in[10], (const float*)d_in[14]};
    const float* bn[3] = {(const float*)d_in[7],  (const float*)d_in[11], (const float*)d_in[15]};
    const float* Wr = (const float*)d_in[16];
    const float* br = (const float*)d_in[17];
    const float* Wl = (const float*)d_in[18];
    const float* bl = (const float*)d_in[19];
    float* out = (float*)d_out;

    char* ws = (char*)d_ws;
    size_t off = 0;
    auto carve = [&](size_t bytes) { void* p = ws + off; off += (bytes + 255) & ~(size_t)255; return p; };
    float* aggp = (float*)carve((size_t)N_NODES * HID * 4);
    float* aggn = (float*)carve((size_t)N_NODES * HID * 4);
    float* degp = (float*)carve((size_t)N_NODES * 4);
    float* degn = (float*)carve((size_t)N_NODES * 4);
    float* hbuf = (float*)carve((size_t)N_NODES * HID * 4);
    unsigned short* Apbf = (unsigned short*)carve((size_t)N_NODES * KPAD * 2);
    unsigned short* Anbf = (unsigned short*)carve((size_t)N_NODES * KPAD * 2);
    unsigned short* hbf  = (unsigned short*)carve((size_t)N_NODES * HID * 2);
    unsigned short* wpbf = (unsigned short*)carve((size_t)HALF * KPAD * 2);
    unsigned short* wnbf = (unsigned short*)carve((size_t)HALF * KPAD * 2);
    unsigned short* wrbf = (unsigned short*)carve((size_t)HID * RK * 2);
    float* g = (float*)carve((size_t)NUM_GRAPHS * HID * 4);

    // one-time packs per launch (deterministic, stateless)
    { int total = HID * RK;
      pack_w_kernel<<<(total + 255) / 256, 256, 0, stream>>>(Wr, wrbf, HID, RK, RK); }
    ginit_kernel<<<(NUM_GRAPHS * HID + 255) / 256, 256, 0, stream>>>(br, g);

    const float* cur = x;
    for (int l = 0; l < 3; ++l) {
        int D = (l == 0) ? FEAT : HID;
        int K = 2 * D;
        hipMemsetAsync(aggp, 0, (size_t)N_NODES * HID * 4, stream);
        hipMemsetAsync(aggn, 0, (size_t)N_NODES * HID * 4, stream);
        hipMemsetAsync(degp, 0, (size_t)N_NODES * 4, stream);
        hipMemsetAsync(degn, 0, (size_t)N_NODES * 4, stream);
        pack_w_kernel<<<(HALF * KPAD + 255) / 256, 256, 0, stream>>>(Wp[l], wpbf, HALF, K, KPAD);
        pack_w_kernel<<<(HALF * KPAD + 255) / 256, 256, 0, stream>>>(Wn[l], wnbf, HALF, K, KPAD);
        deg_kernel<<<(N_EDGES + 255) / 256, 256, 0, stream>>>(ei, ew, degp, degn);
        int chunks = D / 4;
        int total  = N_EDGES * chunks;
        scatter_kernel<<<(total + 255) / 256, 256, 0, stream>>>(cur, ei, ew, aggp, aggn, D, chunks);
        packA_kernel<<<(N_NODES * (KPAD / 2) + 255) / 256, 256, 0, stream>>>(
            aggp, aggn, degp, degn, cur, D, Apbf, Anbf);
        gemm_layer_kernel<<<N_NODES / 16, 256, 0, stream>>>(
            Apbf, Anbf, wpbf, wnbf, bp[l], bn[l], hbuf,
            (l == 2) ? hbf : (unsigned short*)nullptr, (l < 2) ? 1 : 0);
        cur = hbuf;
    }

    dim3 rg(64, 25);  // 64 output tiles x 25 K-chunks of 2048
    readout_gemm_kernel<<<rg, 32, 0, stream>>>(hbf, wrbf, g);
    final_kernel<<<1, 64, 0, stream>>>(g, Wl, bl, out);
}
